// Attention_6983616824415
// MI455X (gfx1250) — matrix operand order repaired and checked by reading, not compiled, not run on hardware
//
#include <hip/hip_runtime.h>
#include <cmath>

typedef __bf16 bf16_t;
typedef __attribute__((ext_vector_type(8)))  __bf16 v8bf;
typedef __attribute__((ext_vector_type(16))) __bf16 v16bf;
typedef __attribute__((ext_vector_type(8)))  float  v8f;
typedef int v4i_ __attribute__((vector_size(16)));   // type expected by async-LDS builtin

#define B_ 4
#define D_ 512
#define S_ 2048
#define L_ 8921

#if __has_builtin(__builtin_amdgcn_global_load_async_to_lds_b128)
#define ASYNC_LDS 1
#else
#define ASYNC_LDS 0
#endif

#if ASYNC_LDS
// 16-byte direct global -> LDS async copy (tracked by ASYNCcnt)
__device__ __forceinline__ void async_b128(const void* g, void* l) {
    __builtin_amdgcn_global_load_async_to_lds_b128(
        (v4i_*)const_cast<void*>(g), (v4i_*)l, 0, 0);
}
__device__ __forceinline__ void wait_async0() {
#if __has_builtin(__builtin_amdgcn_s_wait_asynccnt)
    __builtin_amdgcn_s_wait_asynccnt(0);
#else
    asm volatile("s_wait_asynccnt 0x0" ::: "memory");
#endif
}
#endif

union pk4 { bf16_t h[4]; uint2 u2; };

// ---------------------------------------------------------------------------
// WMMA helpers (wave32, 16x16x32 bf16, f32 accumulate)
// A fragment: tile is row-major [16][32] bf16 in LDS.
//   lane<16  holds row M=lane,    K = {0..7} in v[0..7],  {16..23} in v[8..15]
//   lane>=16 holds row M=lane-16, K = {8..15},            {24..31}
// B fragment: tile staged TRANSPOSED as [N=16][K=32] row-major in LDS.
//   lane<16 holds col N=lane K={0..7},{16..23} ; lane>=16 holds col N=lane-16 K={8..15},{24..31}
// ---------------------------------------------------------------------------
__device__ __forceinline__ v16bf load_a_frag(const bf16_t* tile, int lane) {
    const int r = lane & 15, h = lane >> 4;
    const v8bf* row = (const v8bf*)(tile + r * 32);
    v8bf lo = row[h];
    v8bf hi = row[2 + h];
    return __builtin_shufflevector(lo, hi, 0,1,2,3,4,5,6,7,8,9,10,11,12,13,14,15);
}

__device__ __forceinline__ v16bf load_b_frag(const bf16_t* tileBT, int lane) {
    const int r = lane & 15, h = lane >> 4;
    const v8bf* row = (const v8bf*)(tileBT + r * 32);
    v8bf lo = row[h];
    v8bf hi = row[2 + h];
    return __builtin_shufflevector(lo, hi, 0,1,2,3,4,5,6,7,8,9,10,11,12,13,14,15);
}

__device__ __forceinline__ v8f wmma_bf16(v16bf a, v16bf b, v8f c) {
    return __builtin_amdgcn_wmma_f32_16x16x32_bf16(false, a, false, b, (short)0, c, false, false);
}

__device__ __forceinline__ float elu1(float x) {
    return x > 0.f ? x : (__expf(x) - 1.f);
}

// ---------------------------------------------------------------------------
// Elementwise f32 -> bf16 conversion (used for Q)
// ---------------------------------------------------------------------------
__global__ __launch_bounds__(256) void cvt_kernel(const float* __restrict__ src,
                                                  bf16_t* __restrict__ dst, int n) {
    int i = blockIdx.x * 256 + threadIdx.x;
    if (i < n) dst[i] = (bf16_t)src[i];
}

// ---------------------------------------------------------------------------
// K/V: K[b,s,o] = elu( sum_d Wk[o,d]*H[b,d,s] + bk[o] ), same for V.
// A = H^T (M=s, K=d), B^T[n][k] = Wk[n][k] (row-major slice of Wk).
// Block: 256 thr, tile M=64(s) x N=128(o), both K and V per block.
// Grid: (S/64, D/128, B)
// ---------------------------------------------------------------------------
__global__ __launch_bounds__(256) void kv_kernel(const float* __restrict__ H,
                                                 const float* __restrict__ Wk,
                                                 const float* __restrict__ bk,
                                                 const float* __restrict__ Wv,
                                                 const float* __restrict__ bv,
                                                 bf16_t* __restrict__ Kbf,
                                                 bf16_t* __restrict__ Vbf) {
    __shared__ __align__(16) bf16_t tA[64 * 32];     // [m][k]  H^T tile
    __shared__ __align__(16) bf16_t tBk[128 * 32];   // [n][k]  Wk tile (B^T layout)
    __shared__ __align__(16) bf16_t tBv[128 * 32];

    const int tid  = threadIdx.x;
    const int lane = tid & 31;
    const int w    = tid >> 5;
    const int wm   = w & 3;              // 4 M-strips of 16
    const int wn   = w >> 2;             // 2 N-halves of 64
    const int s0   = blockIdx.x * 64;
    const int n0   = blockIdx.y * 128;
    const int b    = blockIdx.z;

    v8f accK[4], accV[4];
    v8f zero = {};
    #pragma unroll
    for (int j = 0; j < 4; ++j) { accK[j] = zero; accV[j] = zero; }

    for (int kc = 0; kc < D_; kc += 32) {
        // prefetch next H chunk (uniform condition, outside staging loops)
        if (kc + 32 < D_) {
            int pk = tid & 31;                 // row of next chunk
            int pm = (tid >> 5) * 8;           // 32B-spaced points across 64 floats
            __builtin_prefetch(&H[((size_t)b * D_ + (kc + 32 + pk)) * S_ + s0 + pm], 0, 3);
        }
        // stage A: tA[m][k] = H[b][kc+k][s0+m]  -- float4 over m, 4x b16 scatter
        #pragma unroll
        for (int i = 0; i < 2; ++i) {
            int t  = tid + i * 256;            // 0..511
            int m4 = (t & 15) * 4;             // 0..60
            int k  = t >> 4;                   // 0..31
            float4 hv = *(const float4*)&H[((size_t)b * D_ + (kc + k)) * S_ + s0 + m4];
            tA[(m4 + 0) * 32 + k] = (bf16_t)hv.x;
            tA[(m4 + 1) * 32 + k] = (bf16_t)hv.y;
            tA[(m4 + 2) * 32 + k] = (bf16_t)hv.z;
            tA[(m4 + 3) * 32 + k] = (bf16_t)hv.w;
        }
        // stage Bk/Bv: tB[n][k] = W[n0+n][kc+k] -- float4 over k, b64 store
        #pragma unroll
        for (int i = 0; i < 4; ++i) {
            int t  = tid + i * 256;            // 0..1023
            int n  = t >> 3;                   // 0..127
            int k4 = (t & 7) * 4;              // 0..28
            float4 wk4 = *(const float4*)&Wk[(size_t)(n0 + n) * D_ + kc + k4];
            float4 wv4 = *(const float4*)&Wv[(size_t)(n0 + n) * D_ + kc + k4];
            pk4 a, c;
            a.h[0] = (bf16_t)wk4.x; a.h[1] = (bf16_t)wk4.y;
            a.h[2] = (bf16_t)wk4.z; a.h[3] = (bf16_t)wk4.w;
            c.h[0] = (bf16_t)wv4.x; c.h[1] = (bf16_t)wv4.y;
            c.h[2] = (bf16_t)wv4.z; c.h[3] = (bf16_t)wv4.w;
            *(uint2*)&tBk[n * 32 + k4] = a.u2;
            *(uint2*)&tBv[n * 32 + k4] = c.u2;
        }
        __syncthreads();

        v16bf a = load_a_frag(tA + wm * 16 * 32, lane);
        #pragma unroll
        for (int j = 0; j < 4; ++j) {
            v16bf fbk = load_b_frag(tBk + (wn * 64 + j * 16) * 32, lane);
            accK[j] = wmma_bf16(a, fbk, accK[j]);
            v16bf fbv = load_b_frag(tBv + (wn * 64 + j * 16) * 32, lane);
            accV[j] = wmma_bf16(a, fbv, accV[j]);
        }
        __syncthreads();
    }

    // epilogue: bias + ELU + bf16 store, K/V stored as [b][s][o]
    const int r = lane & 15, h = lane >> 4;
    const int sBase = s0 + wm * 16 + h * 8;
    #pragma unroll
    for (int j = 0; j < 4; ++j) {
        int o = n0 + wn * 64 + j * 16 + r;
        float bko = bk[o], bvo = bv[o];
        #pragma unroll
        for (int i = 0; i < 8; ++i) {
            int s = sBase + i;
            size_t idx = ((size_t)b * S_ + s) * D_ + o;
            Kbf[idx] = (bf16_t)elu1(accK[j][i] + bko);
            Vbf[idx] = (bf16_t)elu1(accV[j][i] + bvo);
        }
    }
}

// ---------------------------------------------------------------------------
// E[b,l,s] = sum_d Q[l,d] * K[b,s,d]   (f32 out, written into A-slot of d_out)
// A = Qbf [L][D], B^T[n][k] = Kbf[b][s0+n][kc+k]  (both direct row-major)
// Both operand tiles are pure bf16 row-major copies -> async global->LDS B128
// copies (ASYNCcnt) with double buffering: copy chunk i+1 while WMMAs consume
// chunk i.  Rows l >= L clamp the source (their output rows are store-guarded).
// Block: 256 thr, tile 128(l) x 128(s). Grid: (ceil(L/128), S/128, B)
// ---------------------------------------------------------------------------
__global__ __launch_bounds__(256) void e_kernel(const bf16_t* __restrict__ Qbf,
                                                const bf16_t* __restrict__ Kbf,
                                                float* __restrict__ Eout) {
    __shared__ __align__(16) bf16_t tA[2][128 * 32];
    __shared__ __align__(16) bf16_t tBT[2][128 * 32];

    const int tid  = threadIdx.x;
    const int lane = tid & 31;
    const int w    = tid >> 5;
    const int wm   = w & 3;              // 4 strips of 32 rows (2 tiles)
    const int wn   = w >> 2;             // 2 halves of 64 cols (4 tiles)
    const int l0   = blockIdx.x * 128;
    const int s0   = blockIdx.y * 128;
    const int b    = blockIdx.z;

    v8f acc[2][4];
    v8f zero = {};
    #pragma unroll
    for (int ii = 0; ii < 2; ++ii)
        #pragma unroll
        for (int j = 0; j < 4; ++j) acc[ii][j] = zero;

#if ASYNC_LDS
    auto issue = [&](int kc, int buf) {
        #pragma unroll
        for (int i = 0; i < 2; ++i) {
            int t = tid + i * 256;           // 0..511 = 128 rows x 4 16B chunks
            int m = t >> 2, q = (t & 3) * 8; // q in bf16 elements (8 = 16B)
            int l = l0 + m; if (l > L_ - 1) l = L_ - 1;
            async_b128(Qbf + (size_t)l * D_ + kc + q,
                       &tA[buf][m * 32 + q]);
            async_b128(Kbf + ((size_t)b * S_ + s0 + m) * D_ + kc + q,
                       &tBT[buf][m * 32 + q]);
        }
    };
    issue(0, 0);
    for (int ic = 0; ic < D_ / 32; ++ic) {
        const int cur = ic & 1;
        wait_async0();           // own chunk-ic copies arrived
        __syncthreads();         // everyone's chunk-ic copies visible
        if (ic + 1 < D_ / 32) issue((ic + 1) * 32, cur ^ 1);
        v16bf a0 = load_a_frag(&tA[cur][(wm * 32) * 32], lane);
        v16bf a1 = load_a_frag(&tA[cur][(wm * 32 + 16) * 32], lane);
        #pragma unroll
        for (int j = 0; j < 4; ++j) {
            v16bf fb = load_b_frag(&tBT[cur][(wn * 64 + j * 16) * 32], lane);
            acc[0][j] = wmma_bf16(a0, fb, acc[0][j]);
            acc[1][j] = wmma_bf16(a1, fb, acc[1][j]);
        }
        // LDS reads for this chunk complete before each wave reaches the next
        // barrier (DScnt waited before WMMA), so the next issue() is safe.
    }
#else
    for (int kc = 0; kc < D_; kc += 32) {
        #pragma unroll
        for (int i = 0; i < 16; ++i) {
            int t = tid + i * 256;
            int m = t >> 5, k = t & 31;
            int l = l0 + m;
            tA[0][m * 32 + k] = (l < L_) ? Qbf[(size_t)l * D_ + kc + k] : (bf16_t)0.f;
            tBT[0][m * 32 + k] = Kbf[((size_t)b * S_ + s0 + m) * D_ + kc + k];
        }
        __syncthreads();
        v16bf a0 = load_a_frag(&tA[0][(wm * 32) * 32], lane);
        v16bf a1 = load_a_frag(&tA[0][(wm * 32 + 16) * 32], lane);
        #pragma unroll
        for (int j = 0; j < 4; ++j) {
            v16bf fb = load_b_frag(&tBT[0][(wn * 64 + j * 16) * 32], lane);
            acc[0][j] = wmma_bf16(a0, fb, acc[0][j]);
            acc[1][j] = wmma_bf16(a1, fb, acc[1][j]);
        }
        __syncthreads();
    }
#endif

    const int r = lane & 15, h = lane >> 4;
    #pragma unroll
    for (int ii = 0; ii < 2; ++ii)
        #pragma unroll
        for (int j = 0; j < 4; ++j)
            #pragma unroll
            for (int i = 0; i < 8; ++i) {
                int l = l0 + wm * 32 + ii * 16 + i + 8 * h;
                if (l < L_) {
                    int s = s0 + wn * 64 + j * 16 + r;
                    Eout[((size_t)b * L_ + l) * S_ + s] = acc[ii][j][i];
                }
            }
}

// ---------------------------------------------------------------------------
// In-place row softmax over S=2048, one block per (b,l) row.
// ---------------------------------------------------------------------------
__global__ __launch_bounds__(256) void softmax_kernel(float* __restrict__ A) {
    __shared__ float red[256];
    float* row = A + (size_t)blockIdx.x * S_;
    const int tid = threadIdx.x;

    float v[8];
    float m = -3.402823466e38f;
    #pragma unroll
    for (int i = 0; i < 8; ++i) { v[i] = row[tid + i * 256]; m = fmaxf(m, v[i]); }
    red[tid] = m; __syncthreads();
    for (int s = 128; s > 0; s >>= 1) {
        if (tid < s) red[tid] = fmaxf(red[tid], red[tid + s]);
        __syncthreads();
    }
    m = red[0]; __syncthreads();

    float sum = 0.f;
    #pragma unroll
    for (int i = 0; i < 8; ++i) { v[i] = __expf(v[i] - m); sum += v[i]; }
    red[tid] = sum; __syncthreads();
    for (int s = 128; s > 0; s >>= 1) {
        if (tid < s) red[tid] += red[tid + s];
        __syncthreads();
    }
    float inv = 1.0f / red[0];
    #pragma unroll
    for (int i = 0; i < 8; ++i) row[tid + i * 256] = v[i] * inv;
}

// ---------------------------------------------------------------------------
// C[b,l,d] = sum_s A[b,l,s] * V[b,s,d]
// A staged f32->bf16; B^T[n][k] = Vbf[b][kc+k][d0+n] (transposed during staging)
// Block: 256 thr, tile 128(l) x 128(d). Grid: (ceil(L/128), D/128, B)
// ---------------------------------------------------------------------------
__global__ __launch_bounds__(256) void c_kernel(const float* __restrict__ A,
                                                const bf16_t* __restrict__ Vbf,
                                                float* __restrict__ C) {
    __shared__ __align__(16) bf16_t tA[128 * 32];
    __shared__ __align__(16) bf16_t tBT[128 * 32];

    const int tid  = threadIdx.x;
    const int lane = tid & 31;
    const int w    = tid >> 5;
    const int wm   = w & 3;
    const int wn   = w >> 2;
    const int l0   = blockIdx.x * 128;
    const int d0   = blockIdx.y * 128;
    const int b    = blockIdx.z;

    v8f acc[2][4];
    v8f zero = {};
    #pragma unroll
    for (int ii = 0; ii < 2; ++ii)
        #pragma unroll
        for (int j = 0; j < 4; ++j) acc[ii][j] = zero;

    for (int kc = 0; kc < S_; kc += 32) {
        // A tile: [m][k] f32->bf16, float4 over k, b64 LDS store
        #pragma unroll
        for (int i = 0; i < 4; ++i) {
            int t  = tid + i * 256;           // 0..1023
            int m  = t >> 3;                  // 0..127
            int k4 = (t & 7) * 4;             // 0..28
            int l  = l0 + m;
            float4 av = (l < L_) ? *(const float4*)&A[((size_t)b * L_ + l) * S_ + kc + k4]
                                 : make_float4(0.f, 0.f, 0.f, 0.f);
            pk4 p;
            p.h[0] = (bf16_t)av.x; p.h[1] = (bf16_t)av.y;
            p.h[2] = (bf16_t)av.z; p.h[3] = (bf16_t)av.w;
            *(uint2*)&tA[m * 32 + k4] = p.u2;
        }
        // B^T tile: 8B bf16 loads over n, 4-way transposed scatter
        #pragma unroll
        for (int i = 0; i < 4; ++i) {
            int t  = tid + i * 256;           // 0..1023
            int kk = t >> 5;                  // 0..31
            int n4 = (t & 31) * 4;            // 0..124
            pk4 p;
            p.u2 = *(const uint2*)&Vbf[((size_t)b * S_ + kc + kk) * D_ + d0 + n4];
            tBT[(n4 + 0) * 32 + kk] = p.h[0];
            tBT[(n4 + 1) * 32 + kk] = p.h[1];
            tBT[(n4 + 2) * 32 + kk] = p.h[2];
            tBT[(n4 + 3) * 32 + kk] = p.h[3];
        }
        __syncthreads();

        v16bf a0 = load_a_frag(tA + (wm * 32) * 32, lane);
        v16bf a1 = load_a_frag(tA + (wm * 32 + 16) * 32, lane);
        #pragma unroll
        for (int j = 0; j < 4; ++j) {
            v16bf fb = load_b_frag(tBT + (wn * 64 + j * 16) * 32, lane);
            acc[0][j] = wmma_bf16(a0, fb, acc[0][j]);
            acc[1][j] = wmma_bf16(a1, fb, acc[1][j]);
        }
        __syncthreads();
    }

    const int r = lane & 15, h = lane >> 4;
    #pragma unroll
    for (int ii = 0; ii < 2; ++ii)
        #pragma unroll
        for (int j = 0; j < 4; ++j)
            #pragma unroll
            for (int i = 0; i < 8; ++i) {
                int l = l0 + wm * 32 + ii * 16 + i + 8 * h;
                if (l < L_) {
                    int d = d0 + wn * 64 + j * 16 + r;
                    C[((size_t)b * L_ + l) * D_ + d] = acc[ii][j][i];
                }
            }
}

// ---------------------------------------------------------------------------
extern "C" void kernel_launch(void* const* d_in, const int* in_sizes, int n_in,
                              void* d_out, int out_size, void* d_ws, size_t ws_size,
                              hipStream_t stream) {
    const float* H  = (const float*)d_in[0];
    const float* Wk = (const float*)d_in[1];
    const float* bk = (const float*)d_in[2];
    const float* Wv = (const float*)d_in[3];
    const float* bv = (const float*)d_in[4];
    const float* Q  = (const float*)d_in[5];

    float* C = (float*)d_out;                       // [B, L, D]
    float* A = C + (size_t)B_ * L_ * D_;            // [B, L, S]

    // workspace layout (bf16): Qbf [L][D], Kbf [B][S][D], Vbf [B][S][D]
    char* ws = (char*)d_ws;
    bf16_t* Qbf = (bf16_t*)ws;
    size_t off = ((size_t)L_ * D_ * sizeof(bf16_t) + 255) & ~(size_t)255;
    bf16_t* Kbf = (bf16_t*)(ws + off);
    bf16_t* Vbf = Kbf + (size_t)B_ * S_ * D_;

    // 1) Q -> bf16
    int nq = L_ * D_;
    cvt_kernel<<<(nq + 255) / 256, 256, 0, stream>>>(Q, Qbf, nq);

    // 2) K/V = elu(W.H + b), bf16, [B][S][D]
    kv_kernel<<<dim3(S_ / 64, D_ / 128, B_), 256, 0, stream>>>(H, Wk, bk, Wv, bv, Kbf, Vbf);

    // 3) E = Q K^T  -> A slot of d_out (f32)
    e_kernel<<<dim3((L_ + 127) / 128, S_ / 128, B_), 256, 0, stream>>>(Qbf, Kbf, A);

    // 4) softmax over s, in place
    softmax_kernel<<<B_ * L_, 256, 0, stream>>>(A);

    // 5) C = A V
    c_kernel<<<dim3((L_ + 127) / 128, D_ / 128, B_), 256, 0, stream>>>(A, Vbf, C);
}
